// NemotronV3Mamba2Mixer_18545668784304
// MI455X (gfx1250) — compile-verified, matrix operand order checked by reading, not run on hardware
//
#include <hip/hip_runtime.h>
#include <hip/hip_bf16.h>

typedef __attribute__((ext_vector_type(16))) __bf16 v16bf;
typedef __attribute__((ext_vector_type(8)))  __bf16 v8bf;
typedef __attribute__((ext_vector_type(8)))  float  v8f;
typedef __attribute__((ext_vector_type(4)))  unsigned v4u;
typedef __attribute__((ext_vector_type(8)))  int      v8i;
typedef __attribute__((ext_vector_type(4)))  int      v4i;

constexpr int HIDN  = 2048;
constexpr int NHEAD = 64;
constexpr int PDIM  = 64;
constexpr int NSTATE= 128;
constexpr int CSZ   = 256;
constexpr int DIN   = 4096;                 // NH*P
constexpr int CDIM  = 4352;                 // DIN + 2*N
constexpr int PROJD = 8512;                 // DIN + CDIM + NH
constexpr int BATCH = 2;
constexpr int SEQ   = 2048;
constexpr int BL    = BATCH * SEQ;          // 4096
constexpr int NCH   = SEQ / CSZ;            // 8

// ---------------------------------------------------------------- WMMA helpers
__device__ __forceinline__ v8f wmma_bf16(v16bf a, v16bf b, v8f c) {
  return __builtin_amdgcn_wmma_f32_16x16x32_bf16(false, a, false, b, (short)0, c,
                                                 false, false);
}

// Fragment load from bf16 row (k-contiguous, LDS or global). hf = lane>>4.
// element j -> K = hf*8 + j ; element 8+j -> K = 16 + hf*8 + j
__device__ __forceinline__ v16bf ld_frag16(const __bf16* rowp, int hf) {
  const __bf16* p = rowp + hf * 8;
  v8bf lo = *(const v8bf*)p;
  v8bf hi = *(const v8bf*)(p + 16);
  return __builtin_shufflevector(lo, hi, 0,1,2,3,4,5,6,7,8,9,10,11,12,13,14,15);
}

// ------------------------------------------- TDM: 2D bf16 tile -> LDS (D# per ISA 8.3/8.4)
// tensor window == tile (tx elements wide, ty rows, row stride = stride elems).
// Destination written contiguous row-major at lds_off.
__device__ __forceinline__ void tdm_load_2d_bf16(const __bf16* gptr, unsigned lds_off,
                                                 unsigned tx, unsigned ty,
                                                 unsigned stride_elems) {
  unsigned long long ga = (unsigned long long)(size_t)gptr;
  v4u g0;
  g0[0] = 1u;                                              // count=1 valid user D#
  g0[1] = lds_off;                                         // lds_addr (bytes)
  g0[2] = (unsigned)(ga & 0xFFFFFFFFu);                    // global_addr[31:0]
  g0[3] = (unsigned)((ga >> 32) & 0x1FFFFFFu) | (2u << 30);// global_addr[56:32] | type=2
  v8i g1;
  g1[0] = (int)(1u << 16);                                 // data_size=1 (2 bytes)
  g1[1] = (int)((tx & 0xFFFFu) << 16);                     // tensor_dim0[15:0]
  g1[2] = (int)(((tx >> 16) & 0xFFFFu) | ((ty & 0xFFFFu) << 16)); // dim0 hi | dim1 lo
  g1[3] = (int)(((ty >> 16) & 0xFFFFu) | ((tx & 0xFFFFu) << 16)); // dim1 hi | tile_dim0
  g1[4] = (int)(ty & 0xFFFFu);                             // tile_dim1 (tile_dim2=0 -> 2D)
  g1[5] = (int)stride_elems;                               // tensor_dim0_stride[31:0]
  g1[6] = 0;                                               // stride0 hi | stride1 lo
  g1[7] = 0;
  v4i z4 = {0, 0, 0, 0};
  v8i z8 = {0, 0, 0, 0, 0, 0, 0, 0};
  __builtin_amdgcn_tensor_load_to_lds(g0, g1, z4, z4, z8, 0);
}

// async global f32 -> LDS (one dword per lane), completed by s_wait_asynccnt
__device__ __forceinline__ void async_f32_to_lds(const float* g, float* lds) {
  unsigned dst = (unsigned)(size_t)lds;                    // LDS byte offset = low 32 bits
  unsigned long long src = (unsigned long long)(size_t)g;
  asm volatile("global_load_async_to_lds_b32 %0, %1, off" :: "v"(dst), "v"(src)
               : "memory");
}

// ------------------------------------------------------------ f32 -> bf16 cvt
__global__ void k_cvt_bf16(const float* __restrict__ in, __bf16* __restrict__ out) {
  size_t i = ((size_t)blockIdx.x * blockDim.x + threadIdx.x) * 4;
#pragma unroll
  for (int j = 0; j < 4; ++j) out[i + j] = (__bf16)in[i + j];
}

// ------------------------------------------------------------ K1/K9: NT GEMM
// C[M,N] = A[M,K] * B[N,K]^T   (bf16 operands, f32 acc), 64x64 tile per wave
__global__ __launch_bounds__(32) void k_gemm_nt(const __bf16* __restrict__ A,
                                                const __bf16* __restrict__ B,
                                                float* __restrict__ C,
                                                int lda, int ldb, int ldc, int kdim) {
  const int m0 = blockIdx.x * 64;
  const int n0 = blockIdx.y * 64;
  const int lane = threadIdx.x;
  const int mrow = lane & 15;
  const int hf = lane >> 4;
  v8f acc[4][4];
#pragma unroll
  for (int i = 0; i < 4; ++i)
#pragma unroll
    for (int j = 0; j < 4; ++j)
#pragma unroll
      for (int r = 0; r < 8; ++r) acc[i][j][r] = 0.f;
  const __bf16* arow[4];
  const __bf16* brow[4];
#pragma unroll
  for (int i = 0; i < 4; ++i) arow[i] = A + (size_t)(m0 + i * 16 + mrow) * lda;
#pragma unroll
  for (int j = 0; j < 4; ++j) brow[j] = B + (size_t)(n0 + j * 16 + mrow) * ldb;
#pragma unroll 2
  for (int k0 = 0; k0 < kdim; k0 += 32) {
    v16bf af[4], bf[4];
#pragma unroll
    for (int i = 0; i < 4; ++i) af[i] = ld_frag16(arow[i] + k0, hf);
#pragma unroll
    for (int j = 0; j < 4; ++j) bf[j] = ld_frag16(brow[j] + k0, hf);
#pragma unroll
    for (int i = 0; i < 4; ++i)
#pragma unroll
      for (int j = 0; j < 4; ++j) acc[i][j] = wmma_bf16(af[i], bf[j], acc[i][j]);
  }
#pragma unroll
  for (int i = 0; i < 4; ++i)
#pragma unroll
    for (int j = 0; j < 4; ++j)
#pragma unroll
      for (int r = 0; r < 8; ++r)
        C[(size_t)(m0 + i * 16 + hf * 8 + r) * ldc + (n0 + j * 16 + mrow)] =
            acc[i][j][r];
}

// ---------------------------------------------------- K2: causal conv + SiLU
__global__ void k_conv(const float* __restrict__ proj, const float* __restrict__ cw,
                       const float* __restrict__ cb, __bf16* __restrict__ xbc) {
  size_t idx = (size_t)blockIdx.x * blockDim.x + threadIdx.x;  // row*CDIM + c
  int c = (int)(idx % CDIM);
  int row = (int)(idx / CDIM);
  int b = row / SEQ, l = row % SEQ;
  float s = cb[c];
#pragma unroll
  for (int k = 0; k < 4; ++k) {
    int ls = l - 3 + k;
    if (ls >= 0)
      s += proj[(size_t)(b * SEQ + ls) * PROJD + DIN + c] * cw[c * 4 + k];
  }
  xbc[idx] = (__bf16)(s / (1.f + __expf(-s)));   // SiLU
}

// ------------------------------------------------ K3: softplus(dt), dA = dt*A
__global__ void k_dt(const float* __restrict__ proj, const float* __restrict__ dt_bias,
                     const float* __restrict__ A_log, float* __restrict__ dts,
                     float* __restrict__ dAv) {
  int idx = blockIdx.x * blockDim.x + threadIdx.x;  // row*NHEAD + h
  int h = idx & (NHEAD - 1);
  int row = idx >> 6;
  float d = proj[(size_t)row * PROJD + DIN + CDIM + h] + dt_bias[h];
  float sp = (d > 20.f) ? d : log1pf(__expf(d));
  dts[idx] = sp;
  dAv[idx] = sp * (-__expf(A_log[h]));
}

// ----------------------------------------- K4: per-chunk inclusive cumsum(dA)
__global__ void k_cumsum(const float* __restrict__ dAv, float* __restrict__ Acum,
                         float* __restrict__ dAl) {
  int idx = blockIdx.x * blockDim.x + threadIdx.x;  // (b,h,c)
  int c = idx & (NCH - 1);
  int h = (idx >> 3) & (NHEAD - 1);
  int b = idx >> 9;
  int base = b * SEQ + c * CSZ;
  float run = 0.f;
  float* outp = Acum + ((size_t)(b * NHEAD + h) * NCH + c) * CSZ;
  for (int s = 0; s < CSZ; ++s) {
    run += dAv[(size_t)(base + s) * NHEAD + h];
    outp[s] = run;
  }
  dAl[(b * NHEAD + h) * NCH + c] = run;
}

// --------------------------- K5: intra-chunk Y_diag (WMMA) + chunk states
__global__ __launch_bounds__(256) void k_chunk(const __bf16* __restrict__ xbc,
                                               const float* __restrict__ dts,
                                               const float* __restrict__ Acum,
                                               float* __restrict__ yb,
                                               float* __restrict__ states) {
  const int h = blockIdx.x, c = blockIdx.y, b = blockIdx.z;
  __shared__ __align__(16) __bf16 Cs[CSZ][NSTATE];   // 64 KB (TDM dest)
  __shared__ __align__(16) __bf16 Bs[CSZ][NSTATE];   // 64 KB (TDM dest)
  __shared__ __align__(16) __bf16 XsT[PDIM][CSZ];    // 32 KB (x*dt, transposed)
  __shared__ __align__(16) __bf16 Gs[8][32][32];     // 16 KB (per-wave masked G)
  __shared__ float Ac[CSZ];
  const int tid = threadIdx.x;
  const int rowbase = b * SEQ + c * CSZ;

  // TDM: B/C tiles (256 rows x 128 bf16, row stride CDIM) straight into LDS
  if (tid < 32) {
    const __bf16* gb = xbc + (size_t)rowbase * CDIM + DIN;
    tdm_load_2d_bf16(gb,          (unsigned)(size_t)&Bs[0][0], NSTATE, CSZ, CDIM);
    tdm_load_2d_bf16(gb + NSTATE, (unsigned)(size_t)&Cs[0][0], NSTATE, CSZ, CDIM);
  }
  // async LDS load of the cumsum vector (one dword per lane)
  async_f32_to_lds(Acum + ((size_t)(b * NHEAD + h) * NCH + c) * CSZ + tid, &Ac[tid]);
  // x*dt, transposed (needs scale, stays on the VALU path)
  for (int i = tid; i < CSZ * PDIM; i += 256) {
    int p = i & (PDIM - 1), s = i >> 6;
    float xv = (float)xbc[(size_t)(rowbase + s) * CDIM + h * PDIM + p];
    XsT[p][s] = (__bf16)(xv * dts[(size_t)(rowbase + s) * NHEAD + h]);
  }
  if (tid < 32) __builtin_amdgcn_s_wait_tensorcnt(0);
  asm volatile("s_wait_asynccnt 0x0" ::: "memory");
  __syncthreads();
  const float dAlast = Ac[CSZ - 1];

  const int wave = tid >> 5, lane = tid & 31;
  const int mrow = lane & 15, hf = lane >> 4;
  const int s0 = wave * 32;              // each wave owns 32 output rows
  v8f acc[2][4];
#pragma unroll
  for (int mi = 0; mi < 2; ++mi)
#pragma unroll
    for (int pi = 0; pi < 4; ++pi)
#pragma unroll
      for (int r = 0; r < 8; ++r) acc[mi][pi][r] = 0.f;

  for (int tb = 0; tb <= wave; ++tb) {   // triangular skip (wave-uniform)
    const int t0 = tb * 32;
    v8f g[2][2];
#pragma unroll
    for (int mi = 0; mi < 2; ++mi)
#pragma unroll
      for (int ti = 0; ti < 2; ++ti)
#pragma unroll
        for (int r = 0; r < 8; ++r) g[mi][ti][r] = 0.f;
    // G = C * B^T  over n (K=128)
#pragma unroll
    for (int kk = 0; kk < 4; ++kk) {
      const int n0 = kk * 32;
      v16bf a0 = ld_frag16(&Cs[s0 + mrow][n0], hf);
      v16bf a1 = ld_frag16(&Cs[s0 + 16 + mrow][n0], hf);
      v16bf b0 = ld_frag16(&Bs[t0 + mrow][n0], hf);
      v16bf b1 = ld_frag16(&Bs[t0 + 16 + mrow][n0], hf);
      g[0][0] = wmma_bf16(a0, b0, g[0][0]);
      g[0][1] = wmma_bf16(a0, b1, g[0][1]);
      g[1][0] = wmma_bf16(a1, b0, g[1][0]);
      g[1][1] = wmma_bf16(a1, b1, g[1][1]);
    }
    // causal decay mask applied on D-layout, restage as bf16 A-fragment
#pragma unroll
    for (int mi = 0; mi < 2; ++mi)
#pragma unroll
      for (int ti = 0; ti < 2; ++ti)
#pragma unroll
        for (int r = 0; r < 8; ++r) {
          int srel = mi * 16 + hf * 8 + r;
          int trel = ti * 16 + mrow;
          int sg = s0 + srel, tg = t0 + trel;
          float v = (sg >= tg) ? g[mi][ti][r] * __expf(Ac[sg] - Ac[tg]) : 0.f;
          Gs[wave][srel][trel] = (__bf16)v;
        }
    asm volatile("s_wait_dscnt 0x0" ::: "memory");  // per-wave LDS RAW
    // Y += Gmask * x_dt   (K = 32 positions of this t-block)
#pragma unroll
    for (int mi = 0; mi < 2; ++mi) {
      v16bf ag = ld_frag16(&Gs[wave][mi * 16 + mrow][0], hf);
#pragma unroll
      for (int pi = 0; pi < 4; ++pi) {
        v16bf bx = ld_frag16(&XsT[pi * 16 + mrow][t0], hf);
        acc[mi][pi] = wmma_bf16(ag, bx, acc[mi][pi]);
      }
    }
  }
  // store Y_diag
#pragma unroll
  for (int mi = 0; mi < 2; ++mi)
#pragma unroll
    for (int pi = 0; pi < 4; ++pi)
#pragma unroll
      for (int r = 0; r < 8; ++r) {
        int sg = s0 + mi * 16 + hf * 8 + r;
        int p = pi * 16 + mrow;
        yb[(size_t)(rowbase + sg) * DIN + h * PDIM + p] = acc[mi][pi][r];
      }
  // end-of-chunk states: states[p,n] = sum_s exp(dAlast-Ac[s]) * xdt[s,p] * B[s,n]
  const int p0 = (tid & 15) * 4;
  const int n0s = (tid >> 4) * 8;
  float sac[4][8];
#pragma unroll
  for (int i = 0; i < 4; ++i)
#pragma unroll
    for (int j = 0; j < 8; ++j) sac[i][j] = 0.f;
  for (int s = 0; s < CSZ; ++s) {
    float wdec = __expf(dAlast - Ac[s]);
    float xw[4], bv[8];
#pragma unroll
    for (int i = 0; i < 4; ++i) xw[i] = (float)XsT[p0 + i][s] * wdec;
#pragma unroll
    for (int j = 0; j < 8; ++j) bv[j] = (float)Bs[s][n0s + j];
#pragma unroll
    for (int i = 0; i < 4; ++i)
#pragma unroll
      for (int j = 0; j < 8; ++j) sac[i][j] += xw[i] * bv[j];
  }
  float* sout = states + ((size_t)(b * NCH + c) * NHEAD + h) * PDIM * NSTATE;
#pragma unroll
  for (int i = 0; i < 4; ++i)
#pragma unroll
    for (int j = 0; j < 8; ++j)
      sout[(size_t)(p0 + i) * NSTATE + n0s + j] = sac[i][j];
}

// --------------------------------------------- K6: inter-chunk state scan
__global__ void k_scan(const float* __restrict__ states, const float* __restrict__ dAl,
                       __bf16* __restrict__ prevs) {
  size_t idx = (size_t)blockIdx.x * blockDim.x + threadIdx.x;
  int n = (int)(idx & (NSTATE - 1));
  int p = (int)((idx >> 7) & (PDIM - 1));
  int h = (int)((idx >> 13) & (NHEAD - 1));
  int b = (int)(idx >> 19);
  float carry = 0.f;
  for (int c = 0; c < NCH; ++c) {
    size_t off = (((size_t)(b * NCH + c) * NHEAD + h) * PDIM + p) * NSTATE + n;
    prevs[off] = (__bf16)carry;
    carry = carry * __expf(dAl[(b * NHEAD + h) * NCH + c]) + states[off];
  }
}

// ----------------- K7: Y_off = C * prev^T (WMMA), fuse exp(Acum) + D*x, RMW y
__global__ __launch_bounds__(256) void k_yoff(const __bf16* __restrict__ xbc,
                                              const __bf16* __restrict__ prevs,
                                              const float* __restrict__ Acum,
                                              const float* __restrict__ Dp,
                                              float* __restrict__ yb) {
  const int h = blockIdx.x, c = blockIdx.y, b = blockIdx.z;
  __shared__ __align__(16) __bf16 Cs[CSZ][NSTATE];   // 64 KB (TDM dest)
  __shared__ __align__(16) __bf16 Ps[PDIM][NSTATE];  // 16 KB (TDM dest)
  __shared__ float Ac[CSZ];
  const int tid = threadIdx.x;
  const int rowbase = b * SEQ + c * CSZ;
  const __bf16* pv = prevs + ((size_t)(b * NCH + c) * NHEAD + h) * PDIM * NSTATE;
  if (tid < 32) {
    tdm_load_2d_bf16(xbc + (size_t)rowbase * CDIM + DIN + NSTATE,
                     (unsigned)(size_t)&Cs[0][0], NSTATE, CSZ, CDIM);
    tdm_load_2d_bf16(pv, (unsigned)(size_t)&Ps[0][0], NSTATE, PDIM, NSTATE);
  }
  async_f32_to_lds(Acum + ((size_t)(b * NHEAD + h) * NCH + c) * CSZ + tid, &Ac[tid]);
  if (tid < 32) __builtin_amdgcn_s_wait_tensorcnt(0);
  asm volatile("s_wait_asynccnt 0x0" ::: "memory");
  __syncthreads();
  const int wave = tid >> 5, lane = tid & 31;
  const int mrow = lane & 15, hf = lane >> 4;
  const int s0 = wave * 32;
  v8f acc[2][4];
#pragma unroll
  for (int mi = 0; mi < 2; ++mi)
#pragma unroll
    for (int pi = 0; pi < 4; ++pi)
#pragma unroll
      for (int r = 0; r < 8; ++r) acc[mi][pi][r] = 0.f;
#pragma unroll
  for (int kk = 0; kk < 4; ++kk) {
    int n0 = kk * 32;
    v16bf a0 = ld_frag16(&Cs[s0 + mrow][n0], hf);
    v16bf a1 = ld_frag16(&Cs[s0 + 16 + mrow][n0], hf);
#pragma unroll
    for (int pi = 0; pi < 4; ++pi) {
      v16bf bp = ld_frag16(&Ps[pi * 16 + mrow][n0], hf);
      acc[0][pi] = wmma_bf16(a0, bp, acc[0][pi]);
      acc[1][pi] = wmma_bf16(a1, bp, acc[1][pi]);
    }
  }
  const float Dh = Dp[h];
#pragma unroll
  for (int mi = 0; mi < 2; ++mi)
#pragma unroll
    for (int pi = 0; pi < 4; ++pi)
#pragma unroll
      for (int r = 0; r < 8; ++r) {
        int sg = s0 + mi * 16 + hf * 8 + r;
        int p = pi * 16 + mrow;
        size_t row = (size_t)(rowbase + sg);
        float xr = (float)xbc[row * CDIM + h * PDIM + p];
        size_t yi = row * DIN + h * PDIM + p;
        yb[yi] += acc[mi][pi][r] * __expf(Ac[sg]) + Dh * xr;
      }
}

// ------------------- K8: gate (SiLU(z)) + RMSNorm over D_INNER, emit bf16 gn
__global__ __launch_bounds__(256) void k_gate_norm(const float* __restrict__ proj,
                                                   const float* __restrict__ norm_w,
                                                   const float* __restrict__ yb,
                                                   __bf16* __restrict__ gn) {
  const int row = blockIdx.x;
  const int tid = threadIdx.x;
  __shared__ float red[256];
  float gl[16];
  float ss = 0.f;
#pragma unroll
  for (int j = 0; j < 16; ++j) {
    int d = j * 256 + tid;
    float z = proj[(size_t)row * PROJD + d];
    float g = yb[(size_t)row * DIN + d] * (z / (1.f + __expf(-z)));
    gl[j] = g;
    ss += g * g;
  }
  red[tid] = ss;
  for (int off = 128; off > 0; off >>= 1) {
    __syncthreads();
    if (tid < off) red[tid] += red[tid + off];
  }
  __syncthreads();
  float scale = rsqrtf(red[0] * (1.f / (float)DIN) + 1e-5f);
#pragma unroll
  for (int j = 0; j < 16; ++j) {
    int d = j * 256 + tid;
    gn[(size_t)row * DIN + d] = (__bf16)(gl[j] * scale * norm_w[d]);
  }
}

// ------------------------------------------------------------------- launch
extern "C" void kernel_launch(void* const* d_in, const int* in_sizes, int n_in,
                              void* d_out, int out_size, void* d_ws, size_t ws_size,
                              hipStream_t stream) {
  const float* hidden  = (const float*)d_in[0];
  const float* W_in    = (const float*)d_in[1];
  const float* conv_w  = (const float*)d_in[2];
  const float* conv_b  = (const float*)d_in[3];
  const float* dt_bias = (const float*)d_in[4];
  const float* A_log   = (const float*)d_in[5];
  const float* D_par   = (const float*)d_in[6];
  const float* norm_w  = (const float*)d_in[7];
  const float* W_out   = (const float*)d_in[8];
  float* out = (float*)d_out;

  char* w = (char*)d_ws;
  auto alloc = [&](size_t bytes) -> char* {
    char* p = w;
    w += (bytes + 255) & ~(size_t)255;
    return p;
  };
  float*  proj   = (float*) alloc((size_t)BL * PROJD * 4);                 // 139.5 MB
  __bf16* xbc    = (__bf16*)alloc((size_t)BL * CDIM * 2);                  //  35.7 MB
  float*  dts    = (float*) alloc((size_t)BL * NHEAD * 4);
  float*  dAv    = (float*) alloc((size_t)BL * NHEAD * 4);
  float*  Acum   = (float*) alloc((size_t)BATCH * NHEAD * NCH * CSZ * 4);
  float*  dAl    = (float*) alloc((size_t)BATCH * NHEAD * NCH * 4);
  float*  states = (float*) alloc((size_t)BATCH * NCH * NHEAD * PDIM * NSTATE * 4);
  __bf16* prevs  = (__bf16*)alloc((size_t)BATCH * NCH * NHEAD * PDIM * NSTATE * 2);
  float*  yb     = (float*) alloc((size_t)BL * DIN * 4);                   //  67.1 MB
  __bf16* hbf    = (__bf16*)alloc((size_t)BL * HIDN * 2);                  //  16.8 MB
  __bf16* Wibf   = (__bf16*)alloc((size_t)PROJD * HIDN * 2);               //  34.9 MB
  __bf16* Wobf   = (__bf16*)alloc((size_t)HIDN * DIN * 2);                 //  16.8 MB
  __bf16* gnbf   = (__bf16*)alloc((size_t)BL * DIN * 2);                   //  33.5 MB

  // 0) one-shot bf16 conversion of GEMM operands
  k_cvt_bf16<<<(BL * HIDN) / 1024, 256, 0, stream>>>(hidden, hbf);
  k_cvt_bf16<<<(PROJD * HIDN) / 1024, 256, 0, stream>>>(W_in, Wibf);
  k_cvt_bf16<<<(HIDN * DIN) / 1024, 256, 0, stream>>>(W_out, Wobf);
  // 1) in-proj: proj = hidden @ W_in^T
  k_gemm_nt<<<dim3(BL / 64, PROJD / 64), 32, 0, stream>>>(hbf, Wibf, proj,
                                                          HIDN, HIDN, PROJD, HIDN);
  // 2) depthwise causal conv + SiLU (emits bf16 xBC)
  k_conv<<<(BL * CDIM) / 256, 256, 0, stream>>>(proj, conv_w, conv_b, xbc);
  // 3) dt softplus, dA
  k_dt<<<(BL * NHEAD) / 256, 256, 0, stream>>>(proj, dt_bias, A_log, dts, dAv);
  // 4) per-chunk cumsum
  k_cumsum<<<(BATCH * NHEAD * NCH) / 256, 256, 0, stream>>>(dAv, Acum, dAl);
  // 5) intra-chunk Y_diag + chunk states (TDM-staged tiles)
  k_chunk<<<dim3(NHEAD, NCH, BATCH), 256, 0, stream>>>(xbc, dts, Acum, yb, states);
  // 6) inter-chunk scan (emits bf16 prev states)
  k_scan<<<(BATCH * NHEAD * PDIM * NSTATE) / 256, 256, 0, stream>>>(states, dAl, prevs);
  // 7) Y_off + skip connection (TDM-staged tiles)
  k_yoff<<<dim3(NHEAD, NCH, BATCH), 256, 0, stream>>>(xbc, prevs, Acum, D_par, yb);
  // 8) gate + RMSNorm -> bf16 gn
  k_gate_norm<<<BL, 256, 0, stream>>>(proj, norm_w, yb, gnbf);
  // 9) out-proj: out = gn @ W_out^T
  k_gemm_nt<<<dim3(BL / 64, HIDN / 64), 32, 0, stream>>>(gnbf, Wobf, out,
                                                         DIN, DIN, HIDN, DIN);
  (void)in_sizes; (void)n_in; (void)out_size; (void)ws_size;
}
